// MoEConvBlock_50311246905797
// MI455X (gfx1250) — compile-verified, matrix operand order checked by reading, not run on hardware
//
#include <hip/hip_runtime.h>
#include <stdint.h>

// ---------------------------------------------------------------------------
// MoE conv block, top-1 routed, f16 WMMA (16x16x32, f32 accum) on gfx1250.
//   B=16, C=256, H=W=64, E=4, HID=128, RH=128, TOP_K=1
// Intermediates in NHWC f16 (channels innermost): WMMA A/B fragments and D
// stores are 128-bit vector memory ops; y1 carries a zero halo so the 3x3
// SAME conv needs no bounds checks. conv2/conv3 are LDS-free. Each wave
// register-blocks 2 M-strips (B-fragment reuse x2 -> 1.5 b128 loads / wmma).
// ---------------------------------------------------------------------------

typedef _Float16 v16h __attribute__((ext_vector_type(16)));
typedef float    v8f  __attribute__((ext_vector_type(8)));
typedef uint32_t v4u  __attribute__((ext_vector_type(4)));

union FragU { v16h h; v4u q[2]; };
union PackU { _Float16 f[8]; v4u q; };

// CDNA5 16-bit A/B fragment: dwords 0..3 = K{hi8..hi8+7}, dwords 4..7 =
// K{16+hi8..23+hi8}; hi8 = 8 for lanes 16..31. K contiguous -> two 16B loads.
__device__ __forceinline__ v16h load_frag(const _Float16* p, int hi8) {
  FragU f;
  f.q[0] = *(const v4u*)(p + hi8);
  f.q[1] = *(const v4u*)(p + 16 + hi8);
  return f.h;
}

__device__ __forceinline__ v4u pack_relu(v8f a) {
  PackU pk;
#pragma unroll
  for (int r = 0; r < 8; ++r) pk.f[r] = (_Float16)fmaxf(a[r], 0.0f);
  return pk.q;
}

// ---------------- workspace layout (bytes) ----------------
#define OFF_POOLED 0u          // 16*256 f32             = 16384
#define OFF_SCALE  16384u      // 16 f32
#define OFF_EIDX   16448u      // 16 i32
#define OFF_W1H    16512u      // 4*128*256 f16          = 262144
#define OFF_W2H    278656u     // 4*128*9*128 f16        = 1179648
#define OFF_W3H    1458304u    // 4*256*128 f16          = 262144
#define OFF_Y1P    1720448u    // 16 * 66*66 * 128 f16   = 17842176 (NHWC + halo)
#define OFF_Y2P    19562624u   // 16 * 64*64 * 128 f16   = 16777216 (NHWC)
// total ~36.3 MB

#define Y1P_IMG   (4356 * 128)   // 66*66 rows of 128 ch
#define Y2P_IMG   (4096 * 128)

// ---------------- router: global average pool ----------------
__global__ __launch_bounds__(256) void pool_kernel(const float* __restrict__ x,
                                                   float* __restrict__ pooled) {
  __shared__ float red[256];
  const float* p = x + (size_t)blockIdx.x * 4096;
  float s = 0.f;
  for (int i = threadIdx.x; i < 4096; i += 256) s += p[i];
  red[threadIdx.x] = s;
  __syncthreads();
  for (int st = 128; st > 0; st >>= 1) {
    if ((int)threadIdx.x < st) red[threadIdx.x] += red[threadIdx.x + st];
    __syncthreads();
  }
  if (threadIdx.x == 0) pooled[blockIdx.x] = red[0] * (1.0f / 4096.0f);
}

// ---------------- router: MLP + softmax + top-1 ----------------
__global__ __launch_bounds__(128) void router_kernel(const float* __restrict__ pooled,
                                                     const float* __restrict__ Wr1,
                                                     const float* __restrict__ br1,
                                                     const float* __restrict__ Wr2,
                                                     const float* __restrict__ br2,
                                                     float* __restrict__ scale,
                                                     int* __restrict__ eidx) {
  __shared__ float hsh[128];
  __shared__ float lsh[4];
  const int t = threadIdx.x;
  for (int b = 0; b < 16; ++b) {
    float acc = br1[t];
    const float* pb = pooled + b * 256;
    const float* wr = Wr1 + t * 256;
    for (int c = 0; c < 256; ++c) acc += pb[c] * wr[c];
    hsh[t] = fmaxf(acc, 0.f);
    __syncthreads();
    if (t < 4) {
      float lg = br2[t];
      const float* w2 = Wr2 + t * 128;
      for (int j = 0; j < 128; ++j) lg += hsh[j] * w2[j];
      lsh[t] = lg;
    }
    __syncthreads();
    if (t == 0) {
      float m = lsh[0]; int best = 0;
      for (int e = 1; e < 4; ++e) if (lsh[e] > m) { m = lsh[e]; best = e; }
      float ssum = 0.f;
      for (int e = 0; e < 4; ++e) ssum += __expf(lsh[e] - m);
      float val = 1.0f / ssum;              // softmax prob of argmax
      scale[b] = val / (val + 1e-9f);       // top-1 mask renormalization
      eidx[b]  = best;
    }
    __syncthreads();
  }
}

// ---------------- weight conversion ----------------
__global__ __launch_bounds__(256) void cvt_f16_kernel(const float* __restrict__ in,
                                                      _Float16* __restrict__ out, int n) {
  int i = blockIdx.x * 256 + threadIdx.x;
  if (i < n) out[i] = (_Float16)in[i];
}

// W2: [e][o][i][ky][kx] -> f16 [e][o][tap][i]  (K = tap*128 + i contiguous)
__global__ __launch_bounds__(256) void reorder_w2_kernel(const float* __restrict__ in,
                                                         _Float16* __restrict__ out) {
  int idx = blockIdx.x * 256 + threadIdx.x;   // < 589824
  int eo  = idx / 1152;
  int rem = idx - eo * 1152;
  int tap = rem >> 7;
  int i   = rem & 127;
  out[idx] = (_Float16)in[eo * 1152 + i * 9 + tap];
}

// ---------------- conv1: 1x1 C=256 -> HID=128, relu -> y1p (NHWC+halo) ------
// LDS-staged (fp32 NCHW input needs convert+transpose). Block = 2 image rows
// (128 px); wave owns M-strips {mgrp*32, mgrp*32+16} and one 64-px row.
__global__ __launch_bounds__(256) void conv1_kernel(const float* __restrict__ x,
                                                    const _Float16* __restrict__ W1h,
                                                    const int* __restrict__ eidx,
                                                    _Float16* __restrict__ y1p) {
  __shared__ _Float16 panel[2][128 * 32];      // [pixel 0..127][K32] ping-pong, 16 KB
  const int ypair = blockIdx.x & 31;
  const int b     = blockIdx.x >> 5;
  const int p0    = ypair * 128;               // 2 rows = 128 pixels
  const int e     = eidx[b];
  const float*    xb = x   + (size_t)b * 256 * 4096;
  const _Float16* A  = W1h + (size_t)e * 128 * 256;
  _Float16*       yb = y1p + (size_t)b * Y1P_IMG;

  const int lane = threadIdx.x & 31;
  const int wave = threadIdx.x >> 5;
  const int mgrp = wave >> 1;                  // 0..3
  const int rsel = wave & 1;                   // which of the 2 rows
  const int yrow = ypair * 2 + rsel;
  const int hi8  = (lane & 16) ? 8 : 0;
  const int ncol = lane & 15;
  const _Float16* Arow0 = A + (size_t)(mgrp * 32 + ncol) * 256;
  const _Float16* Arow1 = Arow0 + 16 * 256;

  auto stage = [&](int k0, int buf) {          // 32ch x 128px, float4 reads
    int c   = (int)threadIdx.x >> 4;           // 0..15
    int pxb = ((int)threadIdx.x & 15) * 4;
#pragma unroll
    for (int h = 0; h < 2; ++h) {
      int cc = c + 16 * h;
      const float* sp = xb + (size_t)(k0 + cc) * 4096 + p0;
#pragma unroll
      for (int g = 0; g < 2; ++g) {
        int px = pxb + 64 * g;
        float4 v = *(const float4*)(sp + px);
        panel[buf][(px + 0) * 32 + cc] = (_Float16)v.x;
        panel[buf][(px + 1) * 32 + cc] = (_Float16)v.y;
        panel[buf][(px + 2) * 32 + cc] = (_Float16)v.z;
        panel[buf][(px + 3) * 32 + cc] = (_Float16)v.w;
      }
    }
  };

  v8f acc0[4] = {}, acc1[4] = {};
  stage(0, 0);
  __syncthreads();
  for (int ks = 0; ks < 8; ++ks) {
    if (ks < 7) stage((ks + 1) * 32, (ks + 1) & 1);
    v16h a0 = load_frag(Arow0 + ks * 32, hi8);
    v16h a1 = load_frag(Arow1 + ks * 32, hi8);
    const _Float16* pb = panel[ks & 1] + rsel * 64 * 32;
#pragma unroll
    for (int j = 0; j < 4; ++j) {
      v16h bf = load_frag(pb + (j * 16 + ncol) * 32, hi8);
      acc0[j] = __builtin_amdgcn_wmma_f32_16x16x32_f16(false, a0, false, bf,
                                                       (short)0, acc0[j], false, false);
      acc1[j] = __builtin_amdgcn_wmma_f32_16x16x32_f16(false, a1, false, bf,
                                                       (short)0, acc1[j], false, false);
    }
    __syncthreads();
  }
#pragma unroll
  for (int j = 0; j < 4; ++j) {
    int xcol = j * 16 + ncol;
    _Float16* dp = yb + (size_t)((yrow + 1) * 66 + xcol + 1) * 128 + mgrp * 32;
    *(v4u*)(dp + hi8)      = pack_relu(acc0[j]);
    *(v4u*)(dp + 16 + hi8) = pack_relu(acc1[j]);
  }
}

// ---------------- conv2: 3x3 SAME HID->HID, relu -> y2p (NHWC) --------------
// LDS-free: halo in y1p makes all tap reads in-bounds; B straight from global,
// each B fragment feeds 2 wmma (two M-strips per wave).
__global__ __launch_bounds__(256) void conv2_kernel(const _Float16* __restrict__ y1p,
                                                    const _Float16* __restrict__ W2h,
                                                    const int* __restrict__ eidx,
                                                    _Float16* __restrict__ y2p) {
  const int ypair = blockIdx.x & 31;
  const int b     = blockIdx.x >> 5;
  const int e     = eidx[b];
  const _Float16* src = y1p + (size_t)b * Y1P_IMG;
  const _Float16* A   = W2h + (size_t)e * 128 * 1152;
  _Float16*       dst = y2p + (size_t)b * Y2P_IMG;

  const int lane = threadIdx.x & 31;
  const int wave = threadIdx.x >> 5;
  const int mgrp = wave >> 1;
  const int yrow = ypair * 2 + (wave & 1);
  const int hi8  = (lane & 16) ? 8 : 0;
  const int ncol = lane & 15;
  const _Float16* Arow0 = A + (size_t)(mgrp * 32 + ncol) * 1152;
  const _Float16* Arow1 = Arow0 + 16 * 1152;

  int colOff[4];
#pragma unroll
  for (int j = 0; j < 4; ++j) colOff[j] = (j * 16 + ncol) * 128;

  v8f acc0[4] = {}, acc1[4] = {};
  for (int tap = 0; tap < 9; ++tap) {
    const int dy = tap / 3 - 1;
    const int dx = tap % 3 - 1;
    const _Float16* rowp = src + (size_t)((yrow + dy + 1) * 66 + dx + 1) * 128;
    __builtin_prefetch(Arow0 + tap * 128 + 128, 0, 1);   // next tap's weights
#pragma unroll
    for (int cb = 0; cb < 4; ++cb) {
      const int k0 = tap * 128 + cb * 32;
      v16h a0 = load_frag(Arow0 + k0, hi8);
      v16h a1 = load_frag(Arow1 + k0, hi8);
#pragma unroll
      for (int j = 0; j < 4; ++j) {
        v16h bf = load_frag(rowp + colOff[j] + cb * 32, hi8);
        acc0[j] = __builtin_amdgcn_wmma_f32_16x16x32_f16(false, a0, false, bf,
                                                         (short)0, acc0[j], false, false);
        acc1[j] = __builtin_amdgcn_wmma_f32_16x16x32_f16(false, a1, false, bf,
                                                         (short)0, acc1[j], false, false);
      }
    }
  }
#pragma unroll
  for (int j = 0; j < 4; ++j) {
    int xcol = j * 16 + ncol;
    _Float16* dp = dst + (size_t)(yrow * 64 + xcol) * 128 + mgrp * 32;
    *(v4u*)(dp + hi8)      = pack_relu(acc0[j]);
    *(v4u*)(dp + 16 + hi8) = pack_relu(acc1[j]);
  }
}

// ---------------- conv3: 1x1 HID->C, *scale, +residual -> out (NCHW f32) ----
// Wave owns M-strips {w*16, w*16+128}: one block covers all 256 out channels,
// pixel panel read from L0 once instead of twice.
__global__ __launch_bounds__(256) void conv3_kernel(const _Float16* __restrict__ y2p,
                                                    const _Float16* __restrict__ W3h,
                                                    const float* __restrict__ x,
                                                    const int* __restrict__ eidx,
                                                    const float* __restrict__ scale,
                                                    float* __restrict__ out) {
  const int chunk = blockIdx.x & 63;
  const int b     = blockIdx.x >> 6;
  const int p0    = chunk * 64;
  const int e     = eidx[b];
  const float sc  = scale[b];
  const _Float16* src = y2p + (size_t)b * Y2P_IMG;
  const _Float16* A   = W3h + (size_t)e * 256 * 128;
  const float*    xb  = x   + (size_t)b * 256 * 4096;
  float*          ob  = out + (size_t)b * 256 * 4096;

  const int lane = threadIdx.x & 31;
  const int wave = threadIdx.x >> 5;
  const int m0a  = wave * 16;
  const int m0b  = wave * 16 + 128;
  const int hi8  = (lane & 16) ? 8 : 0;
  const int ncol = lane & 15;
  const _Float16* Arow0 = A + (size_t)(m0a + ncol) * 128;
  const _Float16* Arow1 = A + (size_t)(m0b + ncol) * 128;

  v8f acc0[4] = {}, acc1[4] = {};
#pragma unroll
  for (int ks = 0; ks < 4; ++ks) {
    const int k0 = ks * 32;
    v16h a0 = load_frag(Arow0 + k0, hi8);
    v16h a1 = load_frag(Arow1 + k0, hi8);
#pragma unroll
    for (int j = 0; j < 4; ++j) {
      const _Float16* bp = src + (size_t)(p0 + j * 16 + ncol) * 128 + k0;
      v16h bf = load_frag(bp, hi8);
      acc0[j] = __builtin_amdgcn_wmma_f32_16x16x32_f16(false, a0, false, bf,
                                                       (short)0, acc0[j], false, false);
      acc1[j] = __builtin_amdgcn_wmma_f32_16x16x32_f16(false, a1, false, bf,
                                                       (short)0, acc1[j], false, false);
    }
  }
#pragma unroll
  for (int j = 0; j < 4; ++j) {
    int n = p0 + j * 16 + ncol;
#pragma unroll
    for (int r = 0; r < 8; ++r) {
      size_t o0 = (size_t)(m0a + r + hi8) * 4096 + n;
      size_t o1 = (size_t)(m0b + r + hi8) * 4096 + n;
      ob[o0] = sc * acc0[j][r] + xb[o0];
      ob[o1] = sc * acc1[j][r] + xb[o1];
    }
  }
}

// ---------------------------------------------------------------------------
extern "C" void kernel_launch(void* const* d_in, const int* in_sizes, int n_in,
                              void* d_out, int out_size, void* d_ws, size_t ws_size,
                              hipStream_t stream) {
  (void)in_sizes; (void)n_in; (void)out_size; (void)ws_size;
  const float* x   = (const float*)d_in[0];
  const float* Wr1 = (const float*)d_in[1];
  const float* br1 = (const float*)d_in[2];
  const float* Wr2 = (const float*)d_in[3];
  const float* br2 = (const float*)d_in[4];
  const float* W1  = (const float*)d_in[5];
  const float* W2  = (const float*)d_in[6];
  const float* W3  = (const float*)d_in[7];
  float* out = (float*)d_out;
  char*  ws  = (char*)d_ws;

  float*     pooled = (float*)(ws + OFF_POOLED);
  float*     scale  = (float*)(ws + OFF_SCALE);
  int*       eidx   = (int*)(ws + OFF_EIDX);
  _Float16*  W1h    = (_Float16*)(ws + OFF_W1H);
  _Float16*  W2h    = (_Float16*)(ws + OFF_W2H);
  _Float16*  W3h    = (_Float16*)(ws + OFF_W3H);
  _Float16*  y1p    = (_Float16*)(ws + OFF_Y1P);
  _Float16*  y2p    = (_Float16*)(ws + OFF_Y2P);

  // zero y1p so the 1-pixel halo reads as SAME padding (interior overwritten)
  hipMemsetAsync(ws + OFF_Y1P, 0, (size_t)16 * Y1P_IMG * sizeof(_Float16), stream);

  pool_kernel<<<16 * 256, 256, 0, stream>>>(x, pooled);
  router_kernel<<<1, 128, 0, stream>>>(pooled, Wr1, br1, Wr2, br2, scale, eidx);
  cvt_f16_kernel<<<512, 256, 0, stream>>>(W1, W1h, 131072);
  cvt_f16_kernel<<<512, 256, 0, stream>>>(W3, W3h, 131072);
  reorder_w2_kernel<<<2304, 256, 0, stream>>>(W2, W2h);

  conv1_kernel<<<16 * 32, 256, 0, stream>>>(x, W1h, eidx, y1p);
  conv2_kernel<<<16 * 32, 256, 0, stream>>>(y1p, W2h, eidx, y2p);
  conv3_kernel<<<16 * 64, 256, 0, stream>>>(y2p, W3h, x, eidx, scale, out);
}
